// GraphActor_24721831756516
// MI455X (gfx1250) — matrix operand order repaired and checked by reading, not compiled, not run on hardware
//
#include <hip/hip_runtime.h>

typedef _Float16 v16h __attribute__((ext_vector_type(16)));
typedef _Float16 h8   __attribute__((ext_vector_type(8)));
typedef float    v8f  __attribute__((ext_vector_type(8)));

union V16 { v16h v; h8 h[2]; };

__device__ __forceinline__ unsigned fmap(float x) {
    unsigned u = __float_as_uint(x);
    return (u & 0x80000000u) ? ~u : (u | 0x80000000u);
}
__device__ __forceinline__ float funmap(unsigned u) {
    return (u & 0x80000000u) ? __uint_as_float(u & 0x7FFFFFFFu)
                             : __uint_as_float(~u);
}

// branchless tanh: sign(x) * (1 - e^{-2|x|}) / (1 + e^{-2|x|})
// -> one v_exp_f32 + v_rcp_f32, no EXEC-mask divergence
__device__ __forceinline__ float fast_tanh(float x) {
    float t = __expf(-2.0f * fabsf(x));
    float y = (1.0f - t) * __builtin_amdgcn_rcpf(1.0f + t);
    return copysignf(y, x);
}

// ---------------------------------------------------------------------------
// Generic row linear: out[r,j] = sum_k x[r,k]*W[k,j] + b[j], H=32 outputs
// ---------------------------------------------------------------------------
__global__ void k_rowlin(const float* __restrict__ x, const float* __restrict__ W,
                         const float* __restrict__ b, float* __restrict__ out,
                         int R, int K) {
    int t = blockIdx.x * 256 + threadIdx.x;
    if (t >= R * 32) return;
    int r = t >> 5, j = t & 31;
    float acc = b[j];
    for (int k = 0; k < K; k++) acc += x[r * K + k] * W[k * 32 + j];
    out[t] = acc;
}

// ---------------------------------------------------------------------------
// passenger features scatter-mean accumulation into vehicles
// ---------------------------------------------------------------------------
__global__ void k_pas_scatter(const int* __restrict__ vsnd, const int* __restrict__ prcv,
                              const float* __restrict__ pas_feat,
                              float* __restrict__ pas_sum, float* __restrict__ pas_cnt,
                              int E) {
    int e = blockIdx.x * 256 + threadIdx.x;
    if (e >= E) return;
    int v = vsnd[e], p = prcv[e];
    atomicAdd(&pas_cnt[v], 1.0f);
#pragma unroll 8
    for (int j = 0; j < 32; j++) atomicAdd(&pas_sum[v * 32 + j], pas_feat[p * 32 + j]);
}

__global__ void k_veh_feat(const float* __restrict__ veh_x, const float* __restrict__ W_veh,
                           const float* __restrict__ b_veh, const float* __restrict__ pas_sum,
                           const float* __restrict__ pas_cnt, float* __restrict__ veh_feat,
                           int V) {
    int t = blockIdx.x * 256 + threadIdx.x;
    if (t >= V * 64) return;
    int v = t >> 6, j = t & 63;
    float val;
    if (j < 32) {
        val = b_veh[j];
        for (int k = 0; k < 5; k++) val += veh_x[v * 5 + k] * W_veh[k * 32 + j];
    } else {
        val = pas_sum[v * 32 + (j - 32)] / fmaxf(pas_cnt[v], 1.0f);
    }
    veh_feat[t] = val;
}

// ---------------------------------------------------------------------------
// TripModel: per-edge MLP (tanh) + scatter-mean into src requests
// ---------------------------------------------------------------------------
__global__ void k_trip_edge(const float* __restrict__ req_x, const float* __restrict__ attr,
                            const int* __restrict__ src, const int* __restrict__ dst,
                            const float* __restrict__ W_te, const float* __restrict__ b_te,
                            float* __restrict__ trip_sum, float* __restrict__ trip_cnt,
                            int E) {
    int e = blockIdx.x * 256 + threadIdx.x;
    if (e >= E) return;
    int s = src[e], d = dst[e];
    float x[17];
#pragma unroll
    for (int k = 0; k < 8; k++) x[k] = req_x[s * 8 + k];
#pragma unroll
    for (int k = 0; k < 8; k++) x[8 + k] = req_x[d * 8 + k];
    x[16] = attr[e];
    atomicAdd(&trip_cnt[s], 1.0f);
    for (int j = 0; j < 32; j++) {
        float acc = b_te[j];
#pragma unroll
        for (int k = 0; k < 17; k++) acc += x[k] * W_te[k * 32 + j];
        atomicAdd(&trip_sum[s * 32 + j], fast_tanh(acc));
    }
}

__global__ void k_trip_feat(const float* __restrict__ req_x, const float* __restrict__ trip_sum,
                            const float* __restrict__ trip_cnt, const float* __restrict__ W_tn,
                            const float* __restrict__ b_tn, float* __restrict__ trip_feat,
                            int R) {
    int t = blockIdx.x * 256 + threadIdx.x;
    if (t >= R * 32) return;
    int r = t >> 5, j = t & 31;
    float inv = 1.0f / fmaxf(trip_cnt[r], 1.0f);
    float acc = b_tn[j];
#pragma unroll
    for (int k = 0; k < 8; k++) acc += req_x[r * 8 + k] * W_tn[k * 32 + j];
    for (int k = 0; k < 32; k++) acc += trip_sum[r * 32 + k] * inv * W_tn[(8 + k) * 32 + j];
    trip_feat[t] = acc;
}

// ---------------------------------------------------------------------------
// Pack row-major W[K][N] (f32) into WMMA B-fragment-linear f16:
// Wp[((ks*NT+nt)*32+lane)*16 + e] = W[32*ks + 8*(lane/16) + e%8 + 16*(e/8)][16*nt + lane%16]
// -> each lane loads its v16h fragment as one contiguous 32B read.
// ---------------------------------------------------------------------------
__global__ void k_pack(const float* __restrict__ W, _Float16* __restrict__ Wp,
                       int K, int N) {
    int NT = N >> 4;
    int total = (K >> 5) * NT * 32 * 16;
    int t = blockIdx.x * 256 + threadIdx.x;
    if (t >= total) return;
    int e = t & 15, lane = (t >> 4) & 31, rest = t >> 9;
    int nt = rest % NT, ks = rest / NT;
    int k = ks * 32 + 8 * (lane >> 4) + (e & 7) + 16 * (e >> 3);
    int n = nt * 16 + (lane & 15);
    Wp[t] = (_Float16)W[k * N + n];
}

// ---------------------------------------------------------------------------
// Attention: per 16-edge tile build A=[vdest(64)||r2v_feat(32)] in LDS (f16),
// then hidden = tanh(A @ W_a1[96x96] + b) via WMMA, score = hidden @ W_a2 + b.
// Also scatter per-segment count and running max(score).
// ---------------------------------------------------------------------------
__global__ __launch_bounds__(256) void k_att(
    const float* __restrict__ req_x, const float* __restrict__ veh_x,
    const float* __restrict__ attr, const int* __restrict__ snd,
    const int* __restrict__ rcv, const float* __restrict__ veh_feat,
    const _Float16* __restrict__ wa1p, const float* __restrict__ b_a1,
    const float* __restrict__ W_em, const float* __restrict__ b_em,
    const float* __restrict__ W_a2, const float* __restrict__ b_a2,
    float* __restrict__ rv_score, unsigned* __restrict__ seg_max,
    float* __restrict__ seg_cnt, int E) {
    __shared__ __align__(16) _Float16 Atile[8][16 * 96];
    const int wave = threadIdx.x >> 5, lane = threadIdx.x & 31;
    const int base = blockIdx.x * 128 + wave * 16;
    _Float16* A = Atile[wave];
    const int col = lane & 15, half = lane >> 4;

    if (base < E) {
        const int e = col, ge = base + e;
        if (ge < E) {
            const int rv = rcv[ge];
            if (half == 0) {
                for (int j = 0; j < 48; j++) A[e * 96 + j] = (_Float16)veh_feat[rv * 64 + j];
            } else {
                for (int j = 48; j < 64; j++) A[e * 96 + j] = (_Float16)veh_feat[rv * 64 + j];
                const int sn = snd[ge];
                float x[15];
#pragma unroll
                for (int k = 0; k < 8; k++) x[k] = req_x[sn * 8 + k];
#pragma unroll
                for (int k = 0; k < 5; k++) x[8 + k] = veh_x[rv * 5 + k];
                x[13] = attr[ge * 2 + 0];
                x[14] = attr[ge * 2 + 1];
                for (int j = 0; j < 32; j++) {
                    float acc = b_em[j];
#pragma unroll
                    for (int k = 0; k < 15; k++) acc += x[k] * W_em[k * 32 + j];
                    A[e * 96 + 64 + j] = (_Float16)acc;
                }
            }
        } else {
            if (half == 0) for (int j = 0; j < 48; j++) A[e * 96 + j] = (_Float16)0.f;
            else           for (int j = 48; j < 96; j++) A[e * 96 + j] = (_Float16)0.f;
        }
    }
    __syncthreads();
    if (base >= E) return;

    // hoist A fragments (reused across all 6 n-tiles)
    V16 a[3];
#pragma unroll
    for (int ks = 0; ks < 3; ks++) {
        const _Float16* ap = A + col * 96 + ks * 32 + half * 8;
        a[ks].h[0] = *(const h8*)ap;
        a[ks].h[1] = *(const h8*)(ap + 16);
    }

    float sacc[8] = {0, 0, 0, 0, 0, 0, 0, 0};
    for (int nt = 0; nt < 6; nt++) {
        const int n = nt * 16 + col;
        const float bb = b_a1[n];
        v8f acc;
#pragma unroll
        for (int r = 0; r < 8; r++) acc[r] = bb;
#pragma unroll
        for (int ks = 0; ks < 3; ks++) {
            v16h bfrag = *(const v16h*)(wa1p + ((size_t)(ks * 6 + nt) * 32 + lane) * 16);
            acc = __builtin_amdgcn_wmma_f32_16x16x32_f16(false, a[ks].v, false, bfrag,
                                                         (short)0, acc, false, false);
        }
        const float w2 = W_a2[n];
#pragma unroll
        for (int r = 0; r < 8; r++) sacc[r] += fast_tanh(acc[r]) * w2;
    }
    // reduce over the 16 columns held by each half-wave
    for (int m = 1; m < 16; m <<= 1)
#pragma unroll
        for (int r = 0; r < 8; r++) sacc[r] += __shfl_xor(sacc[r], m, 32);

    if (col == 0) {
        const float b2 = b_a2[0];
#pragma unroll
        for (int r = 0; r < 8; r++) {
            const int ge = base + r + 8 * half;
            if (ge < E) {
                const float sc = sacc[r] + b2;
                rv_score[ge] = sc;
                const int s = snd[ge];
                atomicAdd(&seg_cnt[s], 1.0f);
                atomicMax(&seg_max[s], fmap(sc));
            }
        }
    }
}

// exp(score - segmax) in place + segment sum
__global__ void k_softmax_e(const int* __restrict__ snd, float* __restrict__ sc,
                            const unsigned* __restrict__ segmax,
                            float* __restrict__ segsum, int E) {
    int e = blockIdx.x * 256 + threadIdx.x;
    if (e >= E) return;
    int s = snd[e];
    float m = funmap(segmax[s]);
    if (!(m > -1e37f && m < 1e37f)) m = 0.f;
    float ex = expf(sc[e] - m);
    sc[e] = ex;
    atomicAdd(&segsum[s], ex);
}

// veh_agg accumulation: sum_e softmax_e * vdest_e (4 threads/edge, 16 comps each)
__global__ void k_vagg(const int* __restrict__ snd, const int* __restrict__ rcv,
                       const float* __restrict__ ex, const float* __restrict__ segsum,
                       const float* __restrict__ veh_feat, float* __restrict__ vagg,
                       int E) {
    int t = blockIdx.x * 256 + threadIdx.x;
    if (t >= E * 4) return;
    int e = t >> 2, c = t & 3;
    int s = snd[e], r = rcv[e];
    float p = ex[e] / segsum[s];
#pragma unroll
    for (int j = c * 16; j < c * 16 + 16; j++)
        atomicAdd(&vagg[s * 64 + j], p * veh_feat[r * 64 + j]);
}

// assemble action_feat = [req_feat | trip_feat | veh_agg/count] -> f16
__global__ void k_actin(const float* __restrict__ req_feat, const float* __restrict__ trip_feat,
                        const float* __restrict__ vagg, const float* __restrict__ segcnt,
                        _Float16* __restrict__ act, int R) {
    int t = blockIdx.x * 256 + threadIdx.x;
    if (t >= R * 128) return;
    int r = t >> 7, j = t & 127;
    float v;
    if (j < 32)       v = req_feat[r * 32 + j];
    else if (j < 64)  v = trip_feat[r * 32 + (j - 32)];
    else              v = vagg[r * 64 + (j - 64)] / fmaxf(segcnt[r], 1.0f);
    act[t] = (_Float16)v;
}

// ---------------------------------------------------------------------------
// Fused action MLP: tanh(X@W_l1) -> LDS, tanh(H1@W_l2) folded into @W_l3 + b.
// 2 waves/block, 16 rows/wave, h1 tile 16x512 f16 in LDS (16KB/wave).
// A-fragments for each layer are hoisted into registers and reused over all
// 32 n-tiles (layer2 keeps 16 fragments = 128 VGPRs resident).
// ---------------------------------------------------------------------------
__global__ __launch_bounds__(64) void k_action(
    const _Float16* __restrict__ act, const _Float16* __restrict__ wl1p,
    const _Float16* __restrict__ wl2p, const float* __restrict__ b_l1,
    const float* __restrict__ b_l2, const float* __restrict__ W_l3,
    const float* __restrict__ b_l3, float* __restrict__ out, int M) {
    __shared__ __align__(16) _Float16 H1[2][16 * 512];
    const int wave = threadIdx.x >> 5, lane = threadIdx.x & 31;
    const int row0 = blockIdx.x * 32 + wave * 16;
    const int col = lane & 15, half = lane >> 4;
    _Float16* h1 = H1[wave];

    if (row0 < M) {
        V16 a1[4];
#pragma unroll
        for (int ks = 0; ks < 4; ks++) {
            const _Float16* ap = act + (size_t)(row0 + col) * 128 + ks * 32 + half * 8;
            a1[ks].h[0] = *(const h8*)ap;
            a1[ks].h[1] = *(const h8*)(ap + 16);
        }
        for (int nt = 0; nt < 32; nt++) {
            const int n = nt * 16 + col;
            const float bb = b_l1[n];
            v8f acc;
#pragma unroll
            for (int r = 0; r < 8; r++) acc[r] = bb;
#pragma unroll
            for (int ks = 0; ks < 4; ks++) {
                v16h bfrag = *(const v16h*)(wl1p + ((size_t)(ks * 32 + nt) * 32 + lane) * 16);
                acc = __builtin_amdgcn_wmma_f32_16x16x32_f16(false, a1[ks].v, false, bfrag,
                                                             (short)0, acc, false, false);
            }
#pragma unroll
            for (int r = 0; r < 8; r++)
                h1[(r + 8 * half) * 512 + n] = (_Float16)fast_tanh(acc[r]);
        }
    }
    __syncthreads();
    if (row0 >= M) return;

    // hoist all 16 layer-2 A fragments (16x512 row held as 256 halves/lane)
    V16 a2[16];
#pragma unroll
    for (int ks = 0; ks < 16; ks++) {
        const _Float16* ap = h1 + col * 512 + ks * 32 + half * 8;
        a2[ks].h[0] = *(const h8*)ap;
        a2[ks].h[1] = *(const h8*)(ap + 16);
    }

    float sacc[8] = {0, 0, 0, 0, 0, 0, 0, 0};
    for (int nt = 0; nt < 32; nt++) {
        const int n = nt * 16 + col;
        const float bb = b_l2[n];
        v8f acc;
#pragma unroll
        for (int r = 0; r < 8; r++) acc[r] = bb;
#pragma unroll
        for (int ks = 0; ks < 16; ks++) {
            v16h bfrag = *(const v16h*)(wl2p + ((size_t)(ks * 32 + nt) * 32 + lane) * 16);
            acc = __builtin_amdgcn_wmma_f32_16x16x32_f16(false, a2[ks].v, false, bfrag,
                                                         (short)0, acc, false, false);
        }
        const float w3 = W_l3[n];
#pragma unroll
        for (int r = 0; r < 8; r++) sacc[r] += fast_tanh(acc[r]) * w3;
    }
    for (int m = 1; m < 16; m <<= 1)
#pragma unroll
        for (int r = 0; r < 8; r++) sacc[r] += __shfl_xor(sacc[r], m, 32);

    if (col == 0) {
        const float b3 = b_l3[0];
#pragma unroll
        for (int r = 0; r < 8; r++) {
            const int row = row0 + r + 8 * half;
            if (row < M) out[row] = sacc[r] + b3;
        }
    }
}

// ---------------------------------------------------------------------------
extern "C" void kernel_launch(void* const* d_in, const int* in_sizes, int n_in,
                              void* d_out, int out_size, void* d_ws, size_t ws_size,
                              hipStream_t stream) {
    const float* pas_x   = (const float*)d_in[0];
    const float* veh_x   = (const float*)d_in[1];
    const float* req_x   = (const float*)d_in[2];
    const float* rr_attr = (const float*)d_in[3];
    const float* rv_attr = (const float*)d_in[4];
    const int* rr_src = (const int*)d_in[5];
    const int* rr_dst = (const int*)d_in[6];
    const int* rv_snd = (const int*)d_in[7];
    const int* rv_rcv = (const int*)d_in[8];
    const int* vp_snd = (const int*)d_in[9];
    const int* vp_rcv = (const int*)d_in[10];
    const float *W_pas = (const float*)d_in[11], *b_pas = (const float*)d_in[12];
    const float *W_veh = (const float*)d_in[13], *b_veh = (const float*)d_in[14];
    const float *W_req = (const float*)d_in[15], *b_req = (const float*)d_in[16];
    const float *W_te  = (const float*)d_in[17], *b_te  = (const float*)d_in[18];
    const float *W_tn  = (const float*)d_in[19], *b_tn  = (const float*)d_in[20];
    const float *W_em  = (const float*)d_in[21], *b_em  = (const float*)d_in[22];
    const float *W_a1  = (const float*)d_in[23], *b_a1  = (const float*)d_in[24];
    const float *W_a2  = (const float*)d_in[25], *b_a2  = (const float*)d_in[26];
    const float *W_l1  = (const float*)d_in[27], *b_l1  = (const float*)d_in[28];
    const float *W_l2  = (const float*)d_in[29], *b_l2  = (const float*)d_in[30];
    const float *W_l3  = (const float*)d_in[31], *b_l3  = (const float*)d_in[32];

    const int NP = in_sizes[0] / 10, NV = in_sizes[1] / 5, NR = in_sizes[2] / 8;
    const int ERR = in_sizes[5], ERV = in_sizes[7], EVP = in_sizes[9];

    char* ws = (char*)d_ws;
    size_t off = 0;
    auto take = [&](size_t bytes) -> char* {
        char* p = ws + off;
        off = (off + bytes + 255) & ~(size_t)255;
        return p;
    };
    float*    pas_feat  = (float*)take((size_t)NP * 32 * 4);
    float*    pas_sum   = (float*)take((size_t)NV * 32 * 4);
    float*    pas_cnt   = (float*)take((size_t)NV * 4);
    float*    veh_feat  = (float*)take((size_t)NV * 64 * 4);
    float*    req_feat  = (float*)take((size_t)NR * 32 * 4);
    float*    trip_sum  = (float*)take((size_t)NR * 32 * 4);
    float*    trip_cnt  = (float*)take((size_t)NR * 4);
    float*    trip_feat = (float*)take((size_t)NR * 32 * 4);
    float*    rv_score  = (float*)take((size_t)ERV * 4);
    unsigned* seg_max   = (unsigned*)take((size_t)NR * 4);
    float*    seg_sum   = (float*)take((size_t)NR * 4);
    float*    seg_cnt   = (float*)take((size_t)NR * 4);
    float*    vagg      = (float*)take((size_t)NR * 64 * 4);
    _Float16* act_in    = (_Float16*)take((size_t)NR * 128 * 2);
    _Float16* wa1p      = (_Float16*)take((size_t)3 * 6 * 32 * 16 * 2);
    _Float16* wl1p      = (_Float16*)take((size_t)4 * 32 * 32 * 16 * 2);
    _Float16* wl2p      = (_Float16*)take((size_t)16 * 32 * 32 * 16 * 2);

    hipMemsetAsync(pas_sum, 0, (size_t)NV * 32 * 4, stream);
    hipMemsetAsync(pas_cnt, 0, (size_t)NV * 4, stream);
    hipMemsetAsync(trip_sum, 0, (size_t)NR * 32 * 4, stream);
    hipMemsetAsync(trip_cnt, 0, (size_t)NR * 4, stream);
    hipMemsetAsync(seg_max, 0, (size_t)NR * 4, stream);
    hipMemsetAsync(seg_sum, 0, (size_t)NR * 4, stream);
    hipMemsetAsync(seg_cnt, 0, (size_t)NR * 4, stream);
    hipMemsetAsync(vagg, 0, (size_t)NR * 64 * 4, stream);

    auto cdiv = [](long a, long b) { return (int)((a + b - 1) / b); };

    k_pack<<<cdiv(3 * 6 * 32 * 16, 256), 256, 0, stream>>>(W_a1, wa1p, 96, 96);
    k_pack<<<cdiv(4 * 32 * 32 * 16, 256), 256, 0, stream>>>(W_l1, wl1p, 128, 512);
    k_pack<<<cdiv(16 * 32 * 32 * 16, 256), 256, 0, stream>>>(W_l2, wl2p, 512, 512);

    k_rowlin<<<cdiv((long)NP * 32, 256), 256, 0, stream>>>(pas_x, W_pas, b_pas, pas_feat, NP, 10);
    k_rowlin<<<cdiv((long)NR * 32, 256), 256, 0, stream>>>(req_x, W_req, b_req, req_feat, NR, 8);
    k_pas_scatter<<<cdiv(EVP, 256), 256, 0, stream>>>(vp_snd, vp_rcv, pas_feat, pas_sum, pas_cnt, EVP);
    k_veh_feat<<<cdiv((long)NV * 64, 256), 256, 0, stream>>>(veh_x, W_veh, b_veh, pas_sum, pas_cnt,
                                                             veh_feat, NV);
    k_trip_edge<<<cdiv(ERR, 256), 256, 0, stream>>>(req_x, rr_attr, rr_src, rr_dst, W_te, b_te,
                                                    trip_sum, trip_cnt, ERR);
    k_trip_feat<<<cdiv((long)NR * 32, 256), 256, 0, stream>>>(req_x, trip_sum, trip_cnt, W_tn, b_tn,
                                                              trip_feat, NR);
    k_att<<<cdiv(ERV, 128), 256, 0, stream>>>(req_x, veh_x, rv_attr, rv_snd, rv_rcv, veh_feat,
                                              wa1p, b_a1, W_em, b_em, W_a2, b_a2,
                                              rv_score, seg_max, seg_cnt, ERV);
    k_softmax_e<<<cdiv(ERV, 256), 256, 0, stream>>>(rv_snd, rv_score, seg_max, seg_sum, ERV);
    k_vagg<<<cdiv((long)ERV * 4, 256), 256, 0, stream>>>(rv_snd, rv_rcv, rv_score, seg_sum,
                                                         veh_feat, vagg, ERV);
    k_actin<<<cdiv((long)NR * 128, 256), 256, 0, stream>>>(req_feat, trip_feat, vagg, seg_cnt,
                                                           act_in, NR);
    k_action<<<cdiv(NR, 32), 64, 0, stream>>>(act_in, wl1p, wl2p, b_l1, b_l2, W_l3, b_l3,
                                              (float*)d_out, NR);
}
